// TopKRouter_89421219103396
// MI455X (gfx1250) — matrix operand order repaired and checked by reading, not compiled, not run on hardware
//
#include <hip/hip_runtime.h>
#include <math.h>

typedef __attribute__((ext_vector_type(16))) __bf16 v16bf;
typedef __attribute__((ext_vector_type(8)))  __bf16 v8bf;
typedef __attribute__((ext_vector_type(8)))  float  v8f;

#define D_DIM   2048
#define E_DIM   64
#define N_TOK   32768
#define CAPACITY 1280.0f

// flattened output layout (all float)
#define OFF_IDX   0          // [N,2] expert indices (as float)
#define OFF_W     65536      // [N,2] expert weights
#define OFF_CNT   131072     // [64]  expert counts
#define OFF_OVF   131136     // scalar capacity_overflow_pct
#define OFF_ZL    131137     // scalar z_loss
#define OFF_ENT   131138     // scalar gate_entropy
#define OFF_VAR   131139     // scalar expert_load_variance
#define OFF_CONF  131140     // [N]   routing_confidence

// workspace layout
#define WS_WB_OFF_BYTES 1024         // bf16 copy of W starts here (64*2048*2 = 256 KB)

__device__ __forceinline__ v16bf cvt16(float4 f0, float4 f1, float4 f2, float4 f3) {
    v16bf r;
    r[0]=(__bf16)f0.x;  r[1]=(__bf16)f0.y;  r[2]=(__bf16)f0.z;  r[3]=(__bf16)f0.w;
    r[4]=(__bf16)f1.x;  r[5]=(__bf16)f1.y;  r[6]=(__bf16)f1.z;  r[7]=(__bf16)f1.w;
    r[8]=(__bf16)f2.x;  r[9]=(__bf16)f2.y;  r[10]=(__bf16)f2.z; r[11]=(__bf16)f2.w;
    r[12]=(__bf16)f3.x; r[13]=(__bf16)f3.y; r[14]=(__bf16)f3.z; r[15]=(__bf16)f3.w;
    return r;
}

// A fragment (16x32 bf16): lane m = lane&15, kbase = (lane>>4)*16.
// Element j = K kbase+j (16 contiguous).
// p points at x[m*D + d0 + kbase]. f32 -> bf16 inline.
__device__ __forceinline__ v16bf load_a_frag(const float* __restrict__ p) {
    const float4* q  = (const float4*)p;
    const float4* q2 = (const float4*)(p + 8);
    return cvt16(q[0], q[1], q2[0], q2[1]);
}

// B fragment (32x16 bf16): lane n = lane&15, kbase = (lane>>4)*16.
// Element j = K kbase+j -> 16 contiguous bf16 at Wb[n*D + d0 + kbase] (32B aligned).
__device__ __forceinline__ v16bf load_b_frag(const __bf16* __restrict__ p) {
    v8bf lo = ((const v8bf*)p)[0];
    v8bf hi = ((const v8bf*)p)[1];
    return __builtin_shufflevector(lo, hi, 0,1,2,3,4,5,6,7,8,9,10,11,12,13,14,15);
}

// Convert gate weights f32 -> bf16 once per launch (vectorized, 4 elems/thread).
__global__ void router_cvt_w(const float* __restrict__ W, __bf16* __restrict__ Wb) {
    const int i = (blockIdx.x * 256 + threadIdx.x) * 4;   // < 64*2048
    float4 f = *(const float4*)(W + i);
    __bf16* o = Wb + i;
    o[0] = (__bf16)f.x; o[1] = (__bf16)f.y; o[2] = (__bf16)f.z; o[3] = (__bf16)f.w;
}

__global__ void router_zero(float* __restrict__ ws) {
    if (threadIdx.x < 66) ws[threadIdx.x] = 0.0f;
}

__global__ __launch_bounds__(256) void router_main(const float* __restrict__ X,
                                                   const __bf16* __restrict__ Wb,
                                                   float* __restrict__ out,
                                                   float* __restrict__ ws) {
    __shared__ float lds[8 * 32 * 64];   // 64 KB: per-wave 32 tokens x 64 experts

    const int wave = threadIdx.x >> 5;
    const int lane = threadIdx.x & 31;
    const int tokenBase = blockIdx.x * 256 + wave * 32;

    const int mrow = lane & 15;
    const int hi   = lane >> 4;     // 0 or 1
    const int akb  = hi * 16;       // A K-base
    const int bkb  = hi * 16;       // B K-base

    const float*  xr0 = X  + (size_t)(tokenBase + mrow)      * D_DIM + akb;
    const float*  xr1 = X  + (size_t)(tokenBase + 16 + mrow) * D_DIM + akb;
    const __bf16* wr0 = Wb + (size_t)(0  + mrow) * D_DIM + bkb;
    const __bf16* wr1 = Wb + (size_t)(16 + mrow) * D_DIM + bkb;
    const __bf16* wr2 = Wb + (size_t)(32 + mrow) * D_DIM + bkb;
    const __bf16* wr3 = Wb + (size_t)(48 + mrow) * D_DIM + bkb;

    v8f acc[2][4];
    v8f z8 = {0.f,0.f,0.f,0.f,0.f,0.f,0.f,0.f};
#pragma unroll
    for (int mt = 0; mt < 2; ++mt)
#pragma unroll
        for (int nt = 0; nt < 4; ++nt) acc[mt][nt] = z8;

    for (int kk = 0; kk < D_DIM / 32; ++kk) {
        const int d0 = kk * 32;
        v16bf a[2], b[4];
        a[0] = load_a_frag(xr0 + d0);
        a[1] = load_a_frag(xr1 + d0);
        b[0] = load_b_frag(wr0 + d0);
        b[1] = load_b_frag(wr1 + d0);
        b[2] = load_b_frag(wr2 + d0);
        b[3] = load_b_frag(wr3 + d0);
#pragma unroll
        for (int mt = 0; mt < 2; ++mt)
#pragma unroll
            for (int nt = 0; nt < 4; ++nt)
                acc[mt][nt] = __builtin_amdgcn_wmma_f32_16x16x32_bf16(
                    false, a[mt], false, b[nt], (short)0, acc[mt][nt], false, false);
    }

    // Spill logits to LDS. C layout: VGPR r, lane L -> M = r + 8*(L>=16), N = L&15.
    float* myLds = lds + wave * (32 * 64);
#pragma unroll
    for (int mt = 0; mt < 2; ++mt)
#pragma unroll
        for (int nt = 0; nt < 4; ++nt)
#pragma unroll
            for (int r = 0; r < 8; ++r) {
                int tl = mt * 16 + r + 8 * hi;
                int e  = nt * 16 + mrow;
                myLds[tl * 64 + e] = acc[mt][nt][r];
            }
    __syncthreads();

    // Phase 2: lane owns local token `lane`. Rotated reads avoid bank conflicts.
    const float* row = myLds + lane * 64;
    float mx = -3.4e38f;
#pragma unroll 8
    for (int i = 0; i < 64; ++i) {
        int e = (lane + i) & 63;
        mx = fmaxf(mx, row[e]);
    }
    float s = 0.f, ulog = 0.f;
    float u1 = -1.f, u2 = -1.f;
    int   i1 = 0,   i2 = 0;
#pragma unroll 4
    for (int i = 0; i < 64; ++i) {
        int e = (lane + i) & 63;
        float l = row[e] - mx;
        float u = __expf(l);
        s += u; ulog += u * l;
        if (u > u1)      { u2 = u1; i2 = i1; u1 = u; i1 = e; }
        else if (u > u2) { u2 = u;  i2 = e; }
    }
    float logs = __logf(s);
    float lse  = mx + logs;                // per-token logsumexp
    float ent  = logs - ulog / s;          // per-token entropy
    float p1 = u1 / s, p2 = u2 / s;
    float wsum = p1 + p2 + 1e-8f;
    float w1 = p1 / wsum, w2 = p2 / wsum;

    const int tg = tokenBase + lane;
    out[OFF_IDX  + 2 * tg]     = (float)i1;
    out[OFF_IDX  + 2 * tg + 1] = (float)i2;
    out[OFF_W    + 2 * tg]     = w1;
    out[OFF_W    + 2 * tg + 1] = w2;
    out[OFF_CONF + tg]         = w1;       // w1 >= w2 -> confidence

    atomicAdd(&ws[i1], 1.0f);
    atomicAdd(&ws[i2], 1.0f);

    // wave-level reduction of z-loss / entropy partials, one atomic per wave
    float rz = lse, re = ent;
#pragma unroll
    for (int o = 16; o > 0; o >>= 1) {
        rz += __shfl_xor(rz, o, 32);
        re += __shfl_xor(re, o, 32);
    }
    if (lane == 0) {
        atomicAdd(&ws[64], rz);
        atomicAdd(&ws[65], re);
    }
}

__global__ void router_finalize(const float* __restrict__ ws, float* __restrict__ out) {
    const int e = threadIdx.x;   // 64 threads
    float c = ws[e];
    out[OFF_CNT + e] = c;
    __syncthreads();
    if (e == 0) {
        float tot = 0.f;
        for (int i = 0; i < 64; ++i) tot += ws[i];
        float ovf = 0.f, var = 0.f;
        const float inv_e = 1.0f / 64.0f;
        for (int i = 0; i < 64; ++i) {
            ovf += fmaxf(ws[i] - CAPACITY, 0.0f);
            float ld = ws[i] / tot - inv_e;
            var += ld * ld;
        }
        out[OFF_OVF] = ovf / (float)N_TOK * 100.0f;
        out[OFF_ZL]  = ws[64] / (float)N_TOK * 0.01f;
        out[OFF_ENT] = ws[65] / (float)N_TOK;
        out[OFF_VAR] = var * inv_e;
    }
}

extern "C" void kernel_launch(void* const* d_in, const int* in_sizes, int n_in,
                              void* d_out, int out_size, void* d_ws, size_t ws_size,
                              hipStream_t stream) {
    const float* X = (const float*)d_in[0];   // hidden_states [4,8192,2048] f32
    const float* W = (const float*)d_in[1];   // gate_weight   [64,2048]     f32
    float*  out = (float*)d_out;
    float*  ws  = (float*)d_ws;
    __bf16* Wb  = (__bf16*)((char*)d_ws + WS_WB_OFF_BYTES);

    router_cvt_w<<<(E_DIM * D_DIM) / (256 * 4), 256, 0, stream>>>(W, Wb);
    router_zero<<<1, 128, 0, stream>>>(ws);
    router_main<<<N_TOK / 256, 256, 0, stream>>>(X, Wb, out, ws);
    router_finalize<<<1, 64, 0, stream>>>(ws, out);
}
